// Model_25812753449473
// MI455X (gfx1250) — matrix operand order repaired and checked by reading, not compiled, not run on hardware
//
#include <hip/hip_runtime.h>

#define B_  256
#define T_  100
#define D_  1024
#define F_  512
#define HW_ 192

typedef __attribute__((ext_vector_type(16))) __bf16 v16bf;
typedef __attribute__((ext_vector_type(8)))  float  v8f;

union Frag16 { v16bf v; unsigned int u[8]; uint4 q[2]; };
union Acc8   { v8f v; float f[8]; };

// f32 -> packed 2x bf16 (RNE). Prefer the hardware packed convert if the
// toolchain exposes it; otherwise fall back to manual round-to-nearest-even.
#if __has_builtin(__builtin_amdgcn_cvt_pk_bf16_f32)
typedef __attribute__((ext_vector_type(2))) __bf16 v2bf;
__device__ __forceinline__ unsigned int pack2(float lo, float hi) {
  union { v2bf v; unsigned int u; } c;
  c.v = __builtin_amdgcn_cvt_pk_bf16_f32(lo, hi);
  return c.u;
}
#else
__device__ __forceinline__ unsigned short f32_bf16_(float x) {
  unsigned int u = __float_as_uint(x);
  u += 0x7FFFu + ((u >> 16) & 1u);
  return (unsigned short)(u >> 16);
}
__device__ __forceinline__ unsigned int pack2(float lo, float hi) {
  return (unsigned int)f32_bf16_(lo) | ((unsigned int)f32_bf16_(hi) << 16);
}
#endif

__device__ __forceinline__ unsigned short f32_bf16(float x) {
  unsigned int u = __float_as_uint(x);
  u += 0x7FFFu + ((u >> 16) & 1u);
  return (unsigned short)(u >> 16);
}

// ------------- one-shot weight conversion: f32 [F,D] -> bf16 [F,D] -----------
__global__ __launch_bounds__(256) void convert_w_kernel(
    const float* __restrict__ Wi, const float* __restrict__ Wt,
    unsigned int* __restrict__ Wi_bf, unsigned int* __restrict__ Wt_bf) {
  const float* src = blockIdx.y ? Wt : Wi;
  unsigned int* dst = blockIdx.y ? Wt_bf : Wi_bf;
  const size_t idx = (size_t)blockIdx.x * 256 + threadIdx.x;   // 8 floats each
  const float4* sp = (const float4*)(src + idx * 8);
  float4 a = sp[0], b = sp[1];
  uint4 o = make_uint4(pack2(a.x, a.y), pack2(a.z, a.w),
                       pack2(b.x, b.y), pack2(b.z, b.w));
  ((uint4*)dst)[idx] = o;
}

// ---------------- image branch: adaptive avg pool -> bf16 [B,D] --------------
__global__ __launch_bounds__(256) void pool_kernel(
    const float* __restrict__ fmv, unsigned short* __restrict__ pooled) {
  const int lane = threadIdx.x & 31;
  const int wv   = threadIdx.x >> 5;
  const int row  = blockIdx.x * 8 + wv;          // (b*D + d)
  const float* p = fmv + (size_t)row * HW_;
  float s = 0.0f;
  #pragma unroll
  for (int j = 0; j < 6; ++j) s += p[lane + j * 32];
  #pragma unroll
  for (int o = 16; o > 0; o >>= 1) s += __shfl_xor(s, o, 32);
  if (lane == 0) pooled[row] = f32_bf16(s * (1.0f / 192.0f));
}

// ---------------- image branch: [16 x D] x [D x 128] WMMA GEMM ---------------
__global__ __launch_bounds__(256) void img_gemm_kernel(
    const unsigned short* __restrict__ pooled, const unsigned short* __restrict__ Wi_bf,
    const float* __restrict__ bi, float* __restrict__ img_raw) {
  const int mt    = blockIdx.x;
  const int fbase = blockIdx.y * 128;
  const int tid   = threadIdx.x;
  const int lane  = tid & 31;
  const int wv    = tid >> 5;
  const int sel   = lane >> 4;
  const int ln    = lane & 15;
  const int f     = fbase + wv * 16 + ln;
  const int row   = mt * 16 + ln;

  const unsigned short* Ap = pooled + (size_t)row * D_;
  const unsigned short* Wp = Wi_bf + (size_t)f * D_ + sel * 16;

  Acc8 acc;
  #pragma unroll
  for (int i = 0; i < 8; ++i) acc.f[i] = 0.0f;

  for (int ks = 0; ks < 32; ++ks) {
    const int kb = ks * 32;
    Frag16 af, bf;
    af.q[0] = *(const uint4*)(Ap + kb + sel * 16);
    af.q[1] = *(const uint4*)(Ap + kb + sel * 16 + 8);
    bf.q[0] = *(const uint4*)(Wp + kb);
    bf.q[1] = *(const uint4*)(Wp + kb + 8);
    acc.v = __builtin_amdgcn_wmma_f32_16x16x32_bf16(
        false, af.v, false, bf.v, (short)0, acc.v, false, false);
  }

  const float bias = bi[f];
  #pragma unroll
  for (int i = 0; i < 8; ++i) {
    const int m = sel * 8 + i;
    img_raw[(mt * 16 + m) * F_ + f] = acc.f[i] + bias;
  }
}

// -------- text branch: fused [112 x D] x [D x 128] WMMA GEMM + masked max ----
__global__ __launch_bounds__(256) void text_gemm_max_kernel(
    const float* __restrict__ S, const unsigned short* __restrict__ Wt_bf,
    const float* __restrict__ bt, const int* __restrict__ len_,
    float* __restrict__ txt_raw) {
  // double-buffered A panel: 112 rows x 32 k, bf16, padded stride 40 elems
  __shared__ __align__(16) unsigned int ldsA[2][112 * 20];   // 17.5 KB

  const int b     = blockIdx.x;
  const int fbase = blockIdx.y * 128;
  const int tid   = threadIdx.x;
  const int lane  = tid & 31;
  const int wv    = tid >> 5;
  const int sel   = lane >> 4;
  const int ln    = lane & 15;
  const int f     = fbase + wv * 16 + ln;
  const int len   = len_[b];

  // A staging: 2 threads per row, each converts 16 f32 -> 16 bf16
  const int  arow  = tid >> 1;                  // rows 0..111 active
  const int  ahalf = tid & 1;
  const bool aact  = (arow < 112);
  const int  srow  = (arow < T_) ? arow : (T_ - 1);   // clamp padded rows
  const float* Sp  = S + (size_t)b * T_ * D_ + (size_t)srow * D_ + ahalf * 16;
  const unsigned short* Wp = Wt_bf + (size_t)f * D_ + sel * 16;

  Acc8 acc[7];
  #pragma unroll
  for (int t = 0; t < 7; ++t)
    #pragma unroll
    for (int i = 0; i < 8; ++i) acc[t].f[i] = 0.0f;

  // software pipeline: prefetch K-step 0
  float4 a0, a1, a2, a3;
  if (aact) {
    const float4* ap = (const float4*)Sp;
    a0 = ap[0]; a1 = ap[1]; a2 = ap[2]; a3 = ap[3];
  }

  for (int ks = 0; ks < 32; ++ks) {
    const int kb  = ks * 32;
    const int buf = ks & 1;

    // issue next K-step's A loads while current step computes
    float4 n0, n1, n2, n3;
    if (aact && ks < 31) {
      const float4* ap = (const float4*)(Sp + kb + 32);
      n0 = ap[0]; n1 = ap[1]; n2 = ap[2]; n3 = ap[3];
    }

    if (aact) {
      unsigned int* dst = &ldsA[buf][arow * 20 + ahalf * 8];
      ((uint4*)dst)[0] = make_uint4(pack2(a0.x, a0.y), pack2(a0.z, a0.w),
                                    pack2(a1.x, a1.y), pack2(a1.z, a1.w));
      ((uint4*)dst)[1] = make_uint4(pack2(a2.x, a2.y), pack2(a2.z, a2.w),
                                    pack2(a3.x, a3.y), pack2(a3.z, a3.w));
    }

    Frag16 bfr;                                 // bf16 weights, direct load (L2)
    bfr.q[0] = *(const uint4*)(Wp + kb);
    bfr.q[1] = *(const uint4*)(Wp + kb + 8);

    __syncthreads();                            // buf[ks&1] tile ready

    #pragma unroll
    for (int tt = 0; tt < 7; ++tt) {            // one B frag reused 7x
      const int row = tt * 16 + ln;
      const unsigned int* src = &ldsA[buf][row * 20 + sel * 8];
      Frag16 af;
      af.q[0] = *(const uint4*)(src);           // k = sel*16 .. sel*16+7
      af.q[1] = *(const uint4*)(src + 4);       // k = sel*16+8 .. sel*16+15
      acc[tt].v = __builtin_amdgcn_wmma_f32_16x16x32_bf16(
          false, af.v, false, bfr.v, (short)0, acc[tt].v, false, false);
    }

    a0 = n0; a1 = n1; a2 = n2; a3 = n3;
  }

  // fused bias + validity-masked max over T (rows t >= len excluded)
  const float bias = bt[f];
  float mx = -3.402823466e38f;
  #pragma unroll
  for (int tt = 0; tt < 7; ++tt)
    #pragma unroll
    for (int i = 0; i < 8; ++i) {
      const int t = tt * 16 + sel * 8 + i;
      const float val = acc[tt].f[i] + bias;
      if (t < len) mx = fmaxf(mx, val);
    }
  mx = fmaxf(mx, __shfl_xor(mx, 16, 32));       // merge the two row-halves
  if (lane < 16) txt_raw[b * F_ + f] = mx;
}

// ---------------- batchnorm over batch axis, both branches -> d_out ----------
__global__ __launch_bounds__(256) void bn_kernel(
    const float* __restrict__ img_raw, const float* __restrict__ txt_raw,
    const float* __restrict__ g_img, const float* __restrict__ be_img,
    const float* __restrict__ g_txt, const float* __restrict__ be_txt,
    float* __restrict__ out) {
  const int br   = blockIdx.y;
  const int lane = threadIdx.x & 31;
  const int wv   = threadIdx.x >> 5;
  const int f    = blockIdx.x * 8 + wv;
  const float* src = br ? txt_raw : img_raw;
  const float  g   = (br ? g_txt  : g_img)[f];
  const float  be  = (br ? be_txt : be_img)[f];

  float x[8], s = 0.0f, sq = 0.0f;
  #pragma unroll
  for (int j = 0; j < 8; ++j) {
    x[j] = src[(lane + j * 32) * F_ + f];
    s += x[j]; sq += x[j] * x[j];
  }
  #pragma unroll
  for (int o = 16; o > 0; o >>= 1) {
    s  += __shfl_xor(s,  o, 32);
    sq += __shfl_xor(sq, o, 32);
  }
  const float mu    = s  * (1.0f / 256.0f);
  const float var   = sq * (1.0f / 256.0f) - mu * mu;   // biased, as torch BN
  const float scale = g * rsqrtf(var + 1e-5f);
  const float shift = be - mu * scale;
  float* o_ = out + (size_t)br * B_ * F_;
  #pragma unroll
  for (int j = 0; j < 8; ++j)
    o_[(lane + j * 32) * F_ + f] = x[j] * scale + shift;
}

extern "C" void kernel_launch(void* const* d_in, const int* in_sizes, int n_in,
                              void* d_out, int out_size, void* d_ws, size_t ws_size,
                              hipStream_t stream) {
  (void)in_sizes; (void)n_in; (void)out_size; (void)ws_size;
  const float* fmv = (const float*)d_in[0];
  const float* S   = (const float*)d_in[1];
  const int*   len = (const int*)  d_in[2];
  const float* Wi  = (const float*)d_in[3];
  const float* bi  = (const float*)d_in[4];
  const float* Wt  = (const float*)d_in[5];
  const float* bt  = (const float*)d_in[6];
  const float* gi  = (const float*)d_in[7];
  const float* bei = (const float*)d_in[8];
  const float* gt  = (const float*)d_in[9];
  const float* bet = (const float*)d_in[10];
  float* out = (float*)d_out;

  char* ws = (char*)d_ws;
  unsigned short* pooled = (unsigned short*)ws;                        // 512 KB
  float* img_raw = (float*)(ws + (size_t)512 * 1024);                  // 512 KB
  float* txt_raw = (float*)(ws + (size_t)1024 * 1024);                 // 512 KB
  unsigned short* Wi_bf = (unsigned short*)(ws + (size_t)1536 * 1024); // 1 MB
  unsigned short* Wt_bf = (unsigned short*)(ws + (size_t)2560 * 1024); // 1 MB

  // F*D / (8 floats per thread * 256 threads) = 256 blocks per matrix
  convert_w_kernel<<<dim3((F_ * D_) / (8 * 256), 2), 256, 0, stream>>>(
      Wi, Wt, (unsigned int*)Wi_bf, (unsigned int*)Wt_bf);
  pool_kernel<<<(B_ * D_) / 8, 256, 0, stream>>>(fmv, pooled);
  img_gemm_kernel<<<dim3(B_ / 16, F_ / 128), 256, 0, stream>>>(pooled, Wi_bf, bi, img_raw);
  text_gemm_max_kernel<<<dim3(B_, F_ / 128), 256, 0, stream>>>(S, Wt_bf, bt, len, txt_raw);
  bn_kernel<<<dim3(F_ / 8, 2), 256, 0, stream>>>(img_raw, txt_raw, gi, bei, gt, bet, out);
}
